// InterpretableMultiHeadAttention_10763188044364
// MI455X (gfx1250) — matrix operand order repaired and checked by reading, not compiled, not run on hardware
//
#include <hip/hip_runtime.h>
#include <hip/hip_bf16.h>

// ---------------- problem constants ----------------
#define BB 8
#define TT 1024
#define DM 1024
#define HH 8
#define DK 128
#define MTOT (BB * TT)                    // 8192 rows for projections
static const size_t OUT_ELEMS = (size_t)BB * TT * DM;          // 8,388,608 (out)
// attn follows out in d_out: (H,B,T,T) fp32

// ---------------- workspace layout (bytes) ----------------
static const size_t OFF_WQT = 0;                                   // (H,128,1024) bf16
static const size_t SZ_WQT  = (size_t)HH * DK * DM * 2;
static const size_t OFF_WKT = OFF_WQT + SZ_WQT;
static const size_t SZ_WKT  = SZ_WQT;
static const size_t OFF_WVT = OFF_WKT + SZ_WKT;                    // (128,1024) bf16
static const size_t SZ_WVT  = (size_t)DK * DM * 2;
static const size_t OFF_WOT = OFF_WVT + SZ_WVT;                    // (1024,128) bf16 = Wo^T
static const size_t SZ_WOT  = (size_t)DM * DK * 2;
static const size_t OFF_QH  = OFF_WOT + SZ_WOT;                    // (H, B*T, 128) bf16
static const size_t SZ_QH   = (size_t)HH * MTOT * DK * 2;
static const size_t OFF_KH  = OFF_QH + SZ_QH;                      // (H, B*T, 128) bf16
static const size_t SZ_KH   = SZ_QH;
static const size_t OFF_VHT = OFF_KH + SZ_KH;                      // (B, 128, 1024) bf16 = vh^T
static const size_t SZ_VHT  = (size_t)BB * DK * TT * 2;
static const size_t OFF_HM  = OFF_VHT + SZ_VHT;                    // (B*T, 128) bf16 mean-head

// ---------------- vector types ----------------
typedef __attribute__((ext_vector_type(16))) __bf16          v16bf;
typedef __attribute__((ext_vector_type(4)))  __bf16          v4bf;
typedef __attribute__((ext_vector_type(8)))  float           v8f;
typedef __attribute__((ext_vector_type(16))) unsigned short  v16u16;
typedef __attribute__((ext_vector_type(8)))  unsigned short  v8u16;
typedef __attribute__((ext_vector_type(4)))  float           v4f;
typedef __attribute__((ext_vector_type(2)))  unsigned int    v2u;

// Native fp32 -> bf16 (v_cvt_pk_bf16_f32, RNE).
__device__ __forceinline__ unsigned short f2bf(float f) {
  __bf16 h = (__bf16)f;
  return __builtin_bit_cast(unsigned short, h);
}

// Convert 4 fp32 -> 4 bf16 and store 8B (used for LDS staging; emits ds_store_b64).
__device__ __forceinline__ void stage4(v4f f, unsigned short* dst) {
  v4bf h;
#pragma unroll
  for (int e = 0; e < 4; ++e) h[e] = (__bf16)f[e];
  *(v2u*)dst = __builtin_bit_cast(v2u, h);
}

// A fragment (16x32 bf16) from bf16 row-major memory (global or LDS row pointer).
__device__ __forceinline__ v16bf load_a_bf16(const unsigned short* row_ptr, int kbase, int half) {
  v8u16 lo = *(const v8u16*)(row_ptr + kbase + half * 16);
  v8u16 hi = *(const v8u16*)(row_ptr + kbase + half * 16 + 8);
  v16u16 r;
#pragma unroll
  for (int e = 0; e < 8; ++e) { r[e] = lo[e]; r[8 + e] = hi[e]; }
  return __builtin_bit_cast(v16bf, r);
}

// B fragment (32x16 bf16): ptr -> BT[n_row] + kc*32 + half*16 (16 contiguous bf16).
__device__ __forceinline__ v16bf load_b_bf16(const unsigned short* ptr) {
  v16u16 r = *(const v16u16*)ptr;
  return __builtin_bit_cast(v16bf, r);
}

__device__ __forceinline__ v8f wmma_bf16(v16bf a, v16bf b, v8f c) {
  return __builtin_amdgcn_wmma_f32_16x16x32_bf16(false, a, false, b, (short)0, c, false, false);
}

// ---------------- kernel 0: fp32 (batch,R,C) -> bf16 (batch,C,R) ----------------
__global__ void k_transpose_bf16(const float* __restrict__ in, unsigned short* __restrict__ out,
                                 int R, int C, int total) {
  int idx = blockIdx.x * blockDim.x + threadIdx.x;
  if (idx >= total) return;
  int c  = idx % C;
  int t  = idx / C;
  int r  = t % R;
  int bt = t / R;
  out[((size_t)bt * C + c) * R + r] = f2bf(in[idx]);
}

// ---------------- kernel 1: projections ----------------
// grid.y: 0 = q -> qh (all 8 heads), 1 = k -> kh (all 8 heads), 2 = v -> vh^T.
// The 16x1024 fp32 activation strip is staged ONCE per block into LDS as bf16;
// all 8 waves build A fragments from LDS (ds_load_b128), B comes from transposed weights.
#define ASTRIDE 1032   // 1024 + 8 ushorts of pad (keeps 8B alignment, spreads banks)
__global__ __launch_bounds__(256) void k_proj(const float* __restrict__ q, const float* __restrict__ k,
                                              const float* __restrict__ v,
                                              const unsigned short* __restrict__ wqT,
                                              const unsigned short* __restrict__ wkT,
                                              const unsigned short* __restrict__ wvT,
                                              unsigned short* __restrict__ qh,
                                              unsigned short* __restrict__ kh,
                                              unsigned short* __restrict__ vhT) {
  __shared__ unsigned short As[16 * ASTRIDE];   // 16 x 1024 bf16 strip (+pad) = ~33 KB
  const int g = blockIdx.y;                      // 0=q, 1=k, 2=v (uniform)
  const float* A = (g == 0) ? q : (g == 1) ? k : v;

  const int lane = threadIdx.x & 31, w = threadIdx.x >> 5;
  const int n = lane & 15, half = lane >> 4;
  const int m0 = blockIdx.x * 16;

  // ---- cooperative stage: 16 rows x 1024 cols fp32 -> bf16 in LDS ----
  {
    const int srow = threadIdx.x >> 4;           // 0..15
    const int ssub = threadIdx.x & 15;           // 0..15
    const float* src = A + (size_t)(m0 + srow) * DM + ssub * 4;
#pragma unroll
    for (int i = 0; i < 16; ++i) {
      v4f f = *(const v4f*)(src + i * 64);       // lanes 0..15 cover 256B contiguous
      stage4(f, &As[srow * ASTRIDE + ssub * 4 + i * 64]);
    }
  }
  __syncthreads();

  if (g < 2) {
    // wave w computes head w: 16 x 128 output, 8 n-tiles, K = 1024.
    const unsigned short* BT  = ((g == 0) ? wqT : wkT) + (size_t)w * DK * DM;
    unsigned short*       OUT = ((g == 0) ? qh  : kh)  + (size_t)w * MTOT * DK;
    v8f acc[8] = {};
    for (int kc = 0; kc < DM / 32; ++kc) {
      v16bf aF = load_a_bf16(&As[n * ASTRIDE], kc * 32, half);   // shared across 8 n-tiles
#pragma unroll
      for (int nt = 0; nt < 8; ++nt) {
        const unsigned short* brow = BT + (size_t)(nt * 16 + n) * DM;
        v16bf bb = load_b_bf16(brow + kc * 32 + half * 16);
        acc[nt] = wmma_bf16(aF, bb, acc[nt]);
      }
    }
    const int mtop = m0 + half * 8;
#pragma unroll
    for (int nt = 0; nt < 8; ++nt) {
      unsigned short* dst = OUT + (size_t)mtop * DK + nt * 16 + n;
#pragma unroll
      for (int r = 0; r < 8; ++r) dst[(size_t)r * DK] = f2bf(acc[nt][r]);
    }
  } else {
    // shared V projection: wave w owns n-tile w; output stored transposed (b, dk, s).
    const unsigned short* brow = wvT + (size_t)(w * 16 + n) * DM;
    v8f acc = {};
    for (int kc = 0; kc < DM / 32; ++kc) {
      v16bf aF = load_a_bf16(&As[n * ASTRIDE], kc * 32, half);
      v16bf bb = load_b_bf16(brow + kc * 32 + half * 16);
      acc = wmma_bf16(aF, bb, acc);
    }
    const int mtop = m0 + half * 8;
    unsigned short* dst = vhT + ((size_t)(m0 >> 10) * DK + w * 16 + n) * TT + (mtop & (TT - 1));
#pragma unroll
    for (int r = 0; r < 8; ++r) dst[r] = f2bf(acc[r]);
  }
}

// ---------------- kernel 2: scores + mask + softmax -> attn (fp32 in d_out) ----------------
__global__ __launch_bounds__(256) void k_scores_softmax(const unsigned short* __restrict__ qh,
                                                        const unsigned short* __restrict__ kh,
                                                        const float* __restrict__ mask,
                                                        float* __restrict__ attn) {
  __shared__ float red[16 * 8];
  __shared__ float rowstat[16];
  const int lane = threadIdx.x & 31, w = threadIdx.x >> 5;
  const int n = lane & 15, half = lane >> 4;
  const int t0 = blockIdx.x * 16, b = blockIdx.y, h = blockIdx.z;
  const size_t hb = (size_t)(h * BB + b);

  const unsigned short* qrow = qh + (hb * TT + t0 + n) * DK;
  const unsigned short* kb   = kh + hb * TT * DK;

  v16bf aF[4];
#pragma unroll
  for (int kc = 0; kc < 4; ++kc) aF[kc] = load_a_bf16(qrow, kc * 32, half);

  const float scale = 0.08838834764831845f;      // 1/sqrt(128)
  float vals[8][8];                               // [s-tile j][row r] register-resident strip
#pragma unroll
  for (int j = 0; j < 8; ++j) {
    const int s0 = (w * 8 + j) * 16;
    const unsigned short* krow = kb + (size_t)(s0 + n) * DK;
    v8f acc = {};
#pragma unroll
    for (int kc = 0; kc < 4; ++kc) {
      v16bf bb = load_b_bf16(krow + kc * 32 + half * 16);
      acc = wmma_bf16(aF[kc], bb, acc);
    }
#pragma unroll
    for (int r = 0; r < 8; ++r) {
      int m = half * 8 + r;
      float mv = mask[((size_t)b * TT + t0 + m) * TT + s0 + n];
      vals[j][r] = acc[r] * scale - 1e9f * (1.0f - mv);
    }
  }

  // row max: per-lane over 8 tiles, then shfl over 16 lanes, then across 8 waves via LDS
  float pmax[8];
#pragma unroll
  for (int r = 0; r < 8; ++r) {
    float m = vals[0][r];
#pragma unroll
    for (int j = 1; j < 8; ++j) m = fmaxf(m, vals[j][r]);
#pragma unroll
    for (int off = 1; off < 16; off <<= 1) m = fmaxf(m, __shfl_xor(m, off, 32));
    pmax[r] = m;
  }
  if (n == 0) {
#pragma unroll
    for (int r = 0; r < 8; ++r) red[(half * 8 + r) * 8 + w] = pmax[r];
  }
  __syncthreads();
  if (threadIdx.x < 16) {
    float m = red[threadIdx.x * 8];
#pragma unroll
    for (int i = 1; i < 8; ++i) m = fmaxf(m, red[threadIdx.x * 8 + i]);
    rowstat[threadIdx.x] = m;
  }
  __syncthreads();
  float rmax[8];
#pragma unroll
  for (int r = 0; r < 8; ++r) rmax[r] = rowstat[half * 8 + r];

  // exp + row sum
  float psum[8];
#pragma unroll
  for (int r = 0; r < 8; ++r) {
    float s = 0.f;
#pragma unroll
    for (int j = 0; j < 8; ++j) {
      float e = __expf(vals[j][r] - rmax[r]);
      vals[j][r] = e;
      s += e;
    }
#pragma unroll
    for (int off = 1; off < 16; off <<= 1) s += __shfl_xor(s, off, 32);
    psum[r] = s;
  }
  __syncthreads();
  if (n == 0) {
#pragma unroll
    for (int r = 0; r < 8; ++r) red[(half * 8 + r) * 8 + w] = psum[r];
  }
  __syncthreads();
  if (threadIdx.x < 16) {
    float s = 0.f;
#pragma unroll
    for (int i = 0; i < 8; ++i) s += red[threadIdx.x * 8 + i];
    rowstat[threadIdx.x] = 1.0f / s;
  }
  __syncthreads();

#pragma unroll
  for (int r = 0; r < 8; ++r) {
    float inv = rowstat[half * 8 + r];
    size_t rowoff = (hb * TT + t0 + half * 8 + r) * (size_t)TT;
#pragma unroll
    for (int j = 0; j < 8; ++j) attn[rowoff + (w * 8 + j) * 16 + n] = vals[j][r] * inv;
  }
}

// ---------------- kernel 3: heads mean = (1/H) * sum_h attn_h @ vh -> hm bf16 ----------------
// attn strip (fp32, from d_out) is staged chunk-wise into LDS as bf16 once per block,
// shared by all 8 waves (wave w owns dk n-tile w).
#define CSTRIDE 264    // 256 + 8 ushorts pad
__global__ __launch_bounds__(256) void k_attn_v(const float* __restrict__ attn,
                                                const unsigned short* __restrict__ vhT,
                                                unsigned short* __restrict__ hm) {
  __shared__ unsigned short Cs[16 * CSTRIDE];   // 16 x 256 bf16 chunk (+pad)
  const int lane = threadIdx.x & 31, w = threadIdx.x >> 5;
  const int n = lane & 15, half = lane >> 4;
  const int t0 = blockIdx.x * 16, b = blockIdx.y;
  const int srow = threadIdx.x >> 4, ssub = threadIdx.x & 15;

  const unsigned short* vrow = vhT + ((size_t)b * DK + w * 16 + n) * TT;
  v8f acc = {};
  for (int h = 0; h < HH; ++h) {
    const float* abase = attn + ((size_t)(h * BB + b) * TT + t0) * TT;
    for (int cc = 0; cc < 4; ++cc) {            // four 256-wide K chunks
      // stage 16 x 256 fp32 -> bf16 into LDS (coalesced 256B reads)
      const float* src = abase + (size_t)srow * TT + cc * 256 + ssub * 4;
#pragma unroll
      for (int i = 0; i < 4; ++i) {
        v4f f = *(const v4f*)(src + i * 64);
        stage4(f, &Cs[srow * CSTRIDE + ssub * 4 + i * 64]);
      }
      __syncthreads();
#pragma unroll
      for (int kc = 0; kc < 8; ++kc) {
        v16bf aF = load_a_bf16(&Cs[n * CSTRIDE], kc * 32, half);
        v16bf bb = load_b_bf16(vrow + (size_t)cc * 256 + kc * 32 + half * 16);
        acc = wmma_bf16(aF, bb, acc);
      }
      __syncthreads();                           // protect LDS before re-stage
    }
  }
  unsigned short* dst = hm + ((size_t)b * TT + t0 + half * 8) * DK + w * 16 + n;
#pragma unroll
  for (int r = 0; r < 8; ++r) dst[(size_t)r * DK] = f2bf(acc[r] * 0.125f);
}

// ---------------- kernel 4: out = hm @ Wo (8192x128x1024) ----------------
__global__ __launch_bounds__(256) void k_out_proj(const unsigned short* __restrict__ hm,
                                                  const unsigned short* __restrict__ woT,
                                                  float* __restrict__ out) {
  const int lane = threadIdx.x & 31, w = threadIdx.x >> 5;
  const int n = lane & 15, half = lane >> 4;
  const int m0 = blockIdx.x * 16;

  const unsigned short* arow = hm + (size_t)(m0 + n) * DK;
  v16bf aF[4];
#pragma unroll
  for (int kc = 0; kc < 4; ++kc) aF[kc] = load_a_bf16(arow, kc * 32, half);

  for (int jt = 0; jt < 8; ++jt) {
    const int n0 = (w + jt * 8) * 16;            // 64 n-tiles over 8 waves
    const unsigned short* brow = woT + (size_t)(n0 + n) * DK;
    v8f acc = {};
#pragma unroll
    for (int kc = 0; kc < 4; ++kc) {
      v16bf bb = load_b_bf16(brow + kc * 32 + half * 16);
      acc = wmma_bf16(aF[kc], bb, acc);
    }
#pragma unroll
    for (int r = 0; r < 8; ++r)
      out[(size_t)(m0 + half * 8 + r) * DM + n0 + n] = acc[r];
  }
}

// ---------------- launch ----------------
extern "C" void kernel_launch(void* const* d_in, const int* in_sizes, int n_in,
                              void* d_out, int out_size, void* d_ws, size_t ws_size,
                              hipStream_t stream) {
  (void)in_sizes; (void)n_in; (void)out_size; (void)ws_size;
  const float* q    = (const float*)d_in[0];
  const float* k    = (const float*)d_in[1];
  const float* v    = (const float*)d_in[2];
  const float* mask = (const float*)d_in[3];
  const float* Wq   = (const float*)d_in[4];
  const float* Wk   = (const float*)d_in[5];
  const float* Wv   = (const float*)d_in[6];
  const float* Wo   = (const float*)d_in[7];

  float* out  = (float*)d_out;
  float* attn = out + OUT_ELEMS;

  char* ws = (char*)d_ws;
  unsigned short* wqT = (unsigned short*)(ws + OFF_WQT);
  unsigned short* wkT = (unsigned short*)(ws + OFF_WKT);
  unsigned short* wvT = (unsigned short*)(ws + OFF_WVT);
  unsigned short* woT = (unsigned short*)(ws + OFF_WOT);
  unsigned short* qh  = (unsigned short*)(ws + OFF_QH);
  unsigned short* kh  = (unsigned short*)(ws + OFF_KH);
  unsigned short* vhT = (unsigned short*)(ws + OFF_VHT);
  unsigned short* hm  = (unsigned short*)(ws + OFF_HM);

  // 0) weight convert + transpose
  {
    int tq = HH * DM * DK;  // 1,048,576
    k_transpose_bf16<<<(tq + 255) / 256, 256, 0, stream>>>(Wq, wqT, DM, DK, tq);
    k_transpose_bf16<<<(tq + 255) / 256, 256, 0, stream>>>(Wk, wkT, DM, DK, tq);
    int tv = DM * DK;       // 131,072
    k_transpose_bf16<<<(tv + 255) / 256, 256, 0, stream>>>(Wv, wvT, DM, DK, tv);
    k_transpose_bf16<<<(tv + 255) / 256, 256, 0, stream>>>(Wo, woT, DK, DM, tv);
  }
  // 1) projections: qh, kh (all heads), vh^T — LDS-staged activations
  k_proj<<<dim3(MTOT / 16, 3), 256, 0, stream>>>(q, k, v, wqT, wkT, wvT, qh, kh, vhT);
  // 2) scores + mask + softmax -> attn (second output)
  k_scores_softmax<<<dim3(TT / 16, BB, HH), 256, 0, stream>>>(qh, kh, mask, attn);
  // 3) attn @ vh (LDS-staged attn chunks), mean over heads -> hm
  k_attn_v<<<dim3(TT / 16, BB), 256, 0, stream>>>(attn, vhT, hm);
  // 4) hm @ Wo -> out (first output)
  k_out_proj<<<MTOT / 16, 256, 0, stream>>>(hm, woT, out);
}
